// Parser_81801947119971
// MI455X (gfx1250) — compile-verified, matrix operand order checked by reading, not run on hardware
//
#include <hip/hip_runtime.h>
#include <hip/hip_bf16.h>

// ---- problem constants ----
#define BB 32
#define WW 128
#define SS 192
#define DD 768
#define HH 10
#define G4 40      // 4*H
#define GN 80      // both directions
#define BH 30
#define CC 50
#define NROW (BB*WW)   // 4096
#define NEGV (-1e30f)

typedef __attribute__((ext_vector_type(16))) _Float16 v16h;
typedef __attribute__((ext_vector_type(8)))  float    v8f;

// ---- workspace layout (bytes, all 256-aligned) ----
#define OFF_BSUM 0u                      // f32 [4096*768]
#define OFF_CNT  12582912u               // f32 [4096]
#define OFF_BH   12599296u               // f16 [4096*768]
#define OFF_WCAT 18890752u               // f16 [80*768]
#define OFF_XG   19013632u               // f32 [4096*80]
#define OFF_UNP  20324352u               // f32 [4096*20]
#define OFF_A1UN 20652032u               // f32 [4096*32]
#define OFF_H1DR 21176320u               // f32 [4096*32]
#define OFF_H2DR 21700608u               // f32 [4096*32]
#define OFF_H2UN 22224896u               // f16 [4096*32]
#define OFF_W1UN 22487040u               // f16 [4096*32]
#define OFF_UNL  22749184u               // f32 [32*128*128]
#define WS_NEEDED 24846336u

__device__ __forceinline__ float sigf(float x) { return 1.0f / (1.0f + expf(-x)); }

// ---- 1) segment accumulate: mean over 3 layers + atomic segment sum ----
__global__ __launch_bounds__(256) void seg_accum(const float* __restrict__ hiddens,
                                                 const int* __restrict__ bert2toks,
                                                 float* __restrict__ bsum,
                                                 float* __restrict__ cnt) {
  int bs = blockIdx.x;                 // b*S + s
  int b = bs / SS, s = bs - b * SS;
  int w = bert2toks[bs];
  const float* h0 = hiddens + ((size_t)(0 * BB + b) * (SS + 1) + (s + 1)) * DD;
  const float* h1 = hiddens + ((size_t)(1 * BB + b) * (SS + 1) + (s + 1)) * DD;
  const float* h2 = hiddens + ((size_t)(2 * BB + b) * (SS + 1) + (s + 1)) * DD;
  float* dst = bsum + (size_t)(b * WW + w) * DD;
  for (int d = threadIdx.x; d < DD; d += 256) {
    float m = (h0[d] + h1[d] + h2[d]) * (1.0f / 3.0f);
    unsafeAtomicAdd(dst + d, m);
  }
  if (threadIdx.x == 0) unsafeAtomicAdd(cnt + b * WW + w, 1.0f);
}

// ---- 2) divide by counts, convert to f16 for WMMA ----
__global__ __launch_bounds__(256) void finalize_bert(const float* __restrict__ bsum,
                                                     const float* __restrict__ cnt,
                                                     _Float16* __restrict__ bh) {
  size_t i = (size_t)blockIdx.x * 256 + threadIdx.x;
  if (i >= (size_t)NROW * DD) return;
  float c = cnt[i / DD];
  bh[i] = (_Float16)(bsum[i] / fmaxf(c, 1.0f));
}

// ---- 3) pack [Wih_f ; Wih_b] (80 x 768) as f16, row-major over K ----
__global__ __launch_bounds__(256) void build_wcat(const float* __restrict__ Wih_f,
                                                  const float* __restrict__ Wih_b,
                                                  _Float16* __restrict__ wcat) {
  int i = blockIdx.x * 256 + threadIdx.x;
  if (i >= GN * DD) return;
  int n = i / DD, k = i - n * DD;
  float v = (n < G4) ? Wih_f[n * DD + k] : Wih_b[(n - G4) * DD + k];
  wcat[i] = (_Float16)v;
}

// ---- 4) xg GEMM: (4096x768)f16 x (768x80)f16 -> f32, + (bih+bhh) ----
// 8 waves/block; each wave owns one 16-row M tile and ALL five 16-col N tiles
// (5 accumulators). The 80x32 B-slab for the current K slice is staged into
// LDS with global_load_async_to_lds_b32 (ASYNCcnt), overlapped with the A
// fragment global load; B fragments then come from LDS (ds_load).
__global__ __launch_bounds__(256) void xg_gemm_lds(const _Float16* __restrict__ A,
                                                   const _Float16* __restrict__ Bm,
                                                   float* __restrict__ xg,
                                                   const float* __restrict__ bih_f,
                                                   const float* __restrict__ bhh_f,
                                                   const float* __restrict__ bih_b,
                                                   const float* __restrict__ bhh_b) {
  __shared__ _Float16 bslab[GN][32];          // 5 KB slab, static LDS (offset 0)
  int tid = threadIdx.x;
  int wid = tid >> 5, lane = tid & 31;
  int m0 = (blockIdx.x * 8 + wid) * 16;
  int r = lane & 15, hi = lane >> 4;
  const _Float16* ap = A + (size_t)(m0 + r) * DD + hi * 16;

  v8f acc[5];
#pragma unroll
  for (int nt = 0; nt < 5; ++nt) acc[nt] = (v8f){};

  for (int ks = 0; ks < 24; ++ks) {
    int k0 = ks * 32;
    __syncthreads();                          // slab reads of prev slice done
    // stage B slab: 80 rows x 32 halves = 1280 dwords, 5 per thread, async
#pragma unroll
    for (int i = 0; i < 5; ++i) {
      int d = tid + i * 256;
      int n = d >> 4, kd = d & 15;
      unsigned ldsoff = n * 64 + kd * 4;      // &bslab[n][kd*2], base 0
      unsigned long long ga =
          (unsigned long long)(uintptr_t)(Bm + (size_t)n * DD + k0 + kd * 2);
      asm volatile("global_load_async_to_lds_b32 %0, %1, off"
                   :: "v"(ldsoff), "v"(ga) : "memory");
    }
    v16h a = *(const v16h*)(ap + k0);         // overlaps the async DMA
    __builtin_prefetch(ap + k0 + 64, 0, 0);   // next A slice -> global_prefetch_b8
    asm volatile("s_wait_asynccnt 0x0" ::: "memory");
    __syncthreads();                          // slab visible to all 8 waves
#pragma unroll
    for (int nt = 0; nt < 5; ++nt) {
      v16h b = *(const v16h*)&bslab[nt * 16 + r][hi * 16];   // ds_load
      acc[nt] = __builtin_amdgcn_wmma_f32_16x16x32_f16(false, a, false, b,
                                                       (short)0, acc[nt],
                                                       false, false);
    }
  }
#pragma unroll
  for (int nt = 0; nt < 5; ++nt) {
    int n = nt * 16 + r;
    float bias = (n < G4) ? (bih_f[n] + bhh_f[n]) : (bih_b[n - G4] + bhh_b[n - G4]);
#pragma unroll
    for (int q = 0; q < 8; ++q) {
      int m = m0 + q + hi * 8;                // ISA C/D layout: VGPR q, lane-hi half
      xg[(size_t)m * GN + n] = acc[nt][q] + bias;
    }
  }
}

// ---- 5) LSTM scan (one block per (batch, direction)) ----
__global__ __launch_bounds__(64) void lstm_kernel(const float* __restrict__ xg,
                                                  const float* __restrict__ Whh_f,
                                                  const float* __restrict__ Whh_b,
                                                  float* __restrict__ unpacked) {
  int b = blockIdx.x, dir = blockIdx.y;
  const float* Whh = (dir == 0) ? Whh_f : Whh_b;
  __shared__ float h[16], c[16], g[G4];
  int tid = threadIdx.x;
  if (tid < HH) { h[tid] = 0.0f; c[tid] = 0.0f; }
  __syncthreads();
  for (int s = 0; s < WW; ++s) {
    int t = (dir == 0) ? s : (WW - 1 - s);
    const float* xr = xg + (size_t)(b * WW + t) * GN + dir * G4;
    if (tid < G4) {
      float acc = xr[tid];                    // xg already holds bih+bhh
#pragma unroll
      for (int j = 0; j < HH; ++j) acc += Whh[tid * HH + j] * h[j];
      g[tid] = acc;
    }
    __syncthreads();
    if (tid < HH) {
      float ig = sigf(g[tid]);
      float fg = sigf(g[HH + tid]);
      float gg = tanhf(g[2 * HH + tid]);
      float og = sigf(g[3 * HH + tid]);
      float cn = fg * c[tid] + ig * gg;
      float hn = og * tanhf(cn);
      c[tid] = cn; h[tid] = hn;
      unpacked[(size_t)(b * WW + t) * (2 * HH) + dir * HH + tid] = hn;
    }
    __syncthreads();
  }
}

// ---- 6) MLP heads: h1/h2 for both biaffines (padded to 32, ones column) ----
__global__ __launch_bounds__(32) void heads_mlp(const float* __restrict__ unpacked,
    const float* un_W1, const float* un_b1, const float* un_W2, const float* un_b2,
    const float* dr_W1, const float* dr_b1, const float* dr_W2, const float* dr_b2,
    float* __restrict__ a1un, float* __restrict__ h1dr, float* __restrict__ h2dr,
    _Float16* __restrict__ h2un) {
  int row = blockIdx.x;
  int j = threadIdx.x;
  __shared__ float x[2 * HH];
  if (j < 2 * HH) x[j] = unpacked[(size_t)row * (2 * HH) + j];
  __syncthreads();
  float v1 = 0.f, v2 = 0.f, v3 = 0.f, v4 = 0.f;
  if (j < BH) {
    float s1 = un_b1[j], s2 = un_b2[j], s3 = dr_b1[j], s4 = dr_b2[j];
#pragma unroll
    for (int hh = 0; hh < 2 * HH; ++hh) {
      float xv = x[hh];
      s1 += xv * un_W1[hh * BH + j];
      s2 += xv * un_W2[hh * BH + j];
      s3 += xv * dr_W1[hh * BH + j];
      s4 += xv * dr_W2[hh * BH + j];
    }
    v1 = fmaxf(s1, 0.f); v2 = fmaxf(s2, 0.f);
    v3 = fmaxf(s3, 0.f); v4 = fmaxf(s4, 0.f);
  } else if (j == BH) {
    v1 = v2 = v3 = v4 = 1.0f;                // ones augmentation
  }
  a1un[(size_t)row * 32 + j] = v1;
  h1dr[(size_t)row * 32 + j] = v3;
  h2dr[(size_t)row * 32 + j] = v4;
  h2un[(size_t)row * 32 + j] = (_Float16)v2;
}

// ---- 7) fold U0 into left factor: w1un[row] = h1a_un[row] @ un_U[0] (31x31) ----
__global__ __launch_bounds__(32) void w1u_kernel(const float* __restrict__ a1un,
                                                 const float* __restrict__ un_U,
                                                 _Float16* __restrict__ w1un) {
  int row = blockIdx.x, g = threadIdx.x;
  __shared__ float a[32];
  a[g] = a1un[(size_t)row * 32 + g];
  __syncthreads();
  float s = 0.f;
  if (g < BH + 1)
    for (int hh = 0; hh < BH + 1; ++hh) s += a[hh] * un_U[hh * (BH + 1) + g];
  w1un[(size_t)row * 32 + g] = (g < BH + 1) ? (_Float16)s : (_Float16)0.f;
}

// ---- 8) unlabeled arc matrix via WMMA: per batch 128x128, K=32 (31 + zero pad) ----
__global__ __launch_bounds__(32) void unlab_gemm(const _Float16* __restrict__ w1,
                                                 const _Float16* __restrict__ h2,
                                                 float* __restrict__ unlab,
                                                 const float* __restrict__ un_bias) {
  int lane = threadIdx.x;
  int b = blockIdx.x >> 6;
  int tile = blockIdx.x & 63;
  int ti = tile >> 3, tj = tile & 7;
  int r = lane & 15, hi = lane >> 4;
  v16h a  = *(const v16h*)(w1 + (size_t)(b * WW + ti * 16 + r) * 32 + hi * 16);
  v16h bb = *(const v16h*)(h2 + (size_t)(b * WW + tj * 16 + r) * 32 + hi * 16);
  v8f acc = {};
  acc = __builtin_amdgcn_wmma_f32_16x16x32_f16(false, a, false, bb, (short)0, acc,
                                               false, false);
  float bias = un_bias[0];
#pragma unroll
  for (int q = 0; q < 8; ++q) {
    int i = ti * 16 + q + hi * 8;
    int j = tj * 16 + r;
    float v = acc[q] + bias;
    if (i == j) v = NEGV;                    // diagonal mask
    unlab[((size_t)(b * WW) + i) * WW + j] = v;
  }
}

// ---- 9) deprel head-gathered scores + CE (ignore==0) ----
__global__ __launch_bounds__(64) void deprel_ce(const float* __restrict__ h1dr,
                                                const float* __restrict__ h2dr,
                                                const float* __restrict__ dr_U,
                                                const float* __restrict__ dr_bias,
                                                const int* __restrict__ heads,
                                                const int* __restrict__ dep_rels,
                                                float* __restrict__ out) {
  int row = blockIdx.x;                      // b*W + i
  int b = row >> 7;
  int tid = threadIdx.x;
  __shared__ float v[32], h1[32], sc[CC];
  if (tid < 32) {
    int head = heads[row];
    v[tid]  = h2dr[(size_t)(b * WW + head) * 32 + tid];
    h1[tid] = h1dr[(size_t)row * 32 + tid];
  }
  __syncthreads();
  if (tid < CC) {
    float s = dr_bias[tid];
    const float* U = dr_U + (size_t)tid * (BH + 1) * (BH + 1);
    for (int hh = 0; hh < BH + 1; ++hh) {
      float ss = 0.f;
#pragma unroll 8
      for (int gg = 0; gg < BH + 1; ++gg) ss += U[hh * (BH + 1) + gg] * v[gg];
      s += h1[hh] * ss;
    }
    sc[tid] = s;
  }
  __syncthreads();
  if (tid == 0) {
    float mx = sc[0];
    for (int c = 1; c < CC; ++c) mx = fmaxf(mx, sc[c]);
    float se = 0.f;
    for (int c = 0; c < CC; ++c) se += expf(sc[c] - mx);
    int tr = dep_rels[row];
    if (tr != 0) unsafeAtomicAdd(out, -(sc[tr] - mx - logf(se)));
  }
}

// ---- 10) unlabeled CE over rows i=1..127, classes j=0..127 ----
__global__ __launch_bounds__(128) void depind_ce(const float* __restrict__ unlab,
                                                 const int* __restrict__ heads,
                                                 const int* __restrict__ masks,
                                                 float* __restrict__ out) {
  int i = blockIdx.x + 1;                    // 1..127
  int b = blockIdx.y;
  int tid = threadIdx.x;
  const float* rowp = unlab + ((size_t)(b * WW) + i) * WW;
  __shared__ float buf[128];
  float v = rowp[tid];
  buf[tid] = v;
  __syncthreads();
  for (int off = 64; off > 0; off >>= 1) {
    if (tid < off) buf[tid] = fmaxf(buf[tid], buf[tid + off]);
    __syncthreads();
  }
  float mx = buf[0];
  __syncthreads();
  buf[tid] = expf(v - mx);
  __syncthreads();
  for (int off = 64; off > 0; off >>= 1) {
    if (tid < off) buf[tid] += buf[tid + off];
    __syncthreads();
  }
  if (tid == 0) {
    float se = buf[0];
    int t = masks[b * WW + i] ? -1 : heads[b * WW + i];
    if (t != -1) unsafeAtomicAdd(out, -(rowp[t] - mx - logf(se)));
  }
}

extern "C" void kernel_launch(void* const* d_in, const int* in_sizes, int n_in,
                              void* d_out, int out_size, void* d_ws, size_t ws_size,
                              hipStream_t stream) {
  if (ws_size < WS_NEEDED) return;
  const float* hiddens   = (const float*)d_in[0];
  const int*   bert2toks = (const int*)d_in[1];
  const int*   heads     = (const int*)d_in[2];
  const int*   dep_rels  = (const int*)d_in[3];
  const int*   masks     = (const int*)d_in[4];
  // d_in[5] = sent_lens (unused by reference math)
  const float* Wih_f = (const float*)d_in[6];
  const float* Whh_f = (const float*)d_in[7];
  const float* bih_f = (const float*)d_in[8];
  const float* bhh_f = (const float*)d_in[9];
  const float* Wih_b = (const float*)d_in[10];
  const float* Whh_b = (const float*)d_in[11];
  const float* bih_b = (const float*)d_in[12];
  const float* bhh_b = (const float*)d_in[13];
  const float* un_W1 = (const float*)d_in[14];
  const float* un_b1 = (const float*)d_in[15];
  const float* un_W2 = (const float*)d_in[16];
  const float* un_b2 = (const float*)d_in[17];
  const float* un_U  = (const float*)d_in[18];
  const float* un_bias = (const float*)d_in[19];
  const float* dr_W1 = (const float*)d_in[20];
  const float* dr_b1 = (const float*)d_in[21];
  const float* dr_W2 = (const float*)d_in[22];
  const float* dr_b2 = (const float*)d_in[23];
  const float* dr_U  = (const float*)d_in[24];
  const float* dr_bias = (const float*)d_in[25];

  char* w = (char*)d_ws;
  float*    bsum = (float*)(w + OFF_BSUM);
  float*    cnt  = (float*)(w + OFF_CNT);
  _Float16* bh   = (_Float16*)(w + OFF_BH);
  _Float16* wcat = (_Float16*)(w + OFF_WCAT);
  float*    xg   = (float*)(w + OFF_XG);
  float*    unp  = (float*)(w + OFF_UNP);
  float*    a1un = (float*)(w + OFF_A1UN);
  float*    h1dr = (float*)(w + OFF_H1DR);
  float*    h2dr = (float*)(w + OFF_H2DR);
  _Float16* h2un = (_Float16*)(w + OFF_H2UN);
  _Float16* w1un = (_Float16*)(w + OFF_W1UN);
  float*    unlab = (float*)(w + OFF_UNL);
  float*    out  = (float*)d_out;

  hipMemsetAsync(bsum, 0, (size_t)OFF_CNT + BB * WW * sizeof(float), stream); // bsum+cnt
  hipMemsetAsync(out, 0, sizeof(float), stream);

  seg_accum<<<BB * SS, 256, 0, stream>>>(hiddens, bert2toks, bsum, cnt);
  build_wcat<<<(GN * DD + 255) / 256, 256, 0, stream>>>(Wih_f, Wih_b, wcat);
  finalize_bert<<<(NROW * DD + 255) / 256, 256, 0, stream>>>(bsum, cnt, bh);
  xg_gemm_lds<<<NROW / (16 * 8), 256, 0, stream>>>(bh, wcat, xg,
                                                   bih_f, bhh_f, bih_b, bhh_b);
  lstm_kernel<<<dim3(BB, 2), 64, 0, stream>>>(xg, Whh_f, Whh_b, unp);
  heads_mlp<<<NROW, 32, 0, stream>>>(unp, un_W1, un_b1, un_W2, un_b2,
                                     dr_W1, dr_b1, dr_W2, dr_b2,
                                     a1un, h1dr, h2dr, h2un);
  w1u_kernel<<<NROW, 32, 0, stream>>>(a1un, un_U, w1un);
  unlab_gemm<<<BB * 64, 32, 0, stream>>>(w1un, h2un, unlab, un_bias);
  deprel_ce<<<NROW, 64, 0, stream>>>(h1dr, h2dr, dr_U, dr_bias, heads, dep_rels, out);
  depind_ce<<<dim3(WW - 1, BB), 128, 0, stream>>>(unlab, heads, masks, out);
}
